// ProposedClusterPredictor_5093831213700
// MI455X (gfx1250) — matrix operand order repaired and checked by reading, not compiled, not run on hardware
//
#include <hip/hip_runtime.h>
#include <hip/hip_bf16.h>
#include <math.h>

typedef float     v8f  __attribute__((ext_vector_type(8)));
typedef float     v2f  __attribute__((ext_vector_type(2)));
typedef _Float16  v16h __attribute__((ext_vector_type(16)));

#if defined(__has_builtin)
#  if __has_builtin(__builtin_amdgcn_wmma_f32_16x16x4_f32)
#    define HAS_F32_WMMA 1
#  endif
#endif
#ifndef HAS_F32_WMMA
#  define HAS_F32_WMMA 0
#endif

#define PB 250  // partial blocks per batch element for x-stats

// ---------------------------------------------------------------------------
// Kernel 1: per-(b,d) streaming stats of x[B][N][64]. Each block handles RPB
// rows of one batch element; float4 loads (256 thr * 16B = 4KB contiguous per
// step). Writes per-block partials [sum(64) | sumsq(64) | max(64)] to ws.
// Deterministic: no float atomics.
// ---------------------------------------------------------------------------
__global__ __launch_bounds__(256) void xstats_kernel(const float* __restrict__ x,
                                                     float* __restrict__ part,
                                                     int N, int RPB) {
  const int pb  = blockIdx.x;
  const int b   = blockIdx.y;
  const int tid = threadIdx.x;
  const int q   = tid & 15;   // feature quad -> d = 4q..4q+3
  const int rg  = tid >> 4;   // row group 0..15
  const int rowStart = pb * RPB;
  int rowEnd = rowStart + RPB; if (rowEnd > N) rowEnd = N;

  float sx=0.f,sy=0.f,sz=0.f,sw=0.f;
  float qx=0.f,qy=0.f,qz=0.f,qw=0.f;
  float mx=-INFINITY,my=-INFINITY,mz=-INFINITY,mw=-INFINITY;

  const float* xb = x + (size_t)b * (size_t)N * 64 + 4 * q;
  for (int n = rowStart + rg; n < rowEnd; n += 16) {
    float4 v = *(const float4*)(xb + (size_t)n * 64);
    sx += v.x; sy += v.y; sz += v.z; sw += v.w;
    qx += v.x*v.x; qy += v.y*v.y; qz += v.z*v.z; qw += v.w*v.w;
    mx = fmaxf(mx, v.x); my = fmaxf(my, v.y); mz = fmaxf(mz, v.z); mw = fmaxf(mw, v.w);
  }

  __shared__ float4 rS[256], rQ[256], rM[256];
  rS[tid] = make_float4(sx, sy, sz, sw);
  rQ[tid] = make_float4(qx, qy, qz, qw);
  rM[tid] = make_float4(mx, my, mz, mw);
  __syncthreads();

  if (tid < 16) {
    float4 S = rS[tid], Q = rQ[tid], M = rM[tid];
    for (int g = 1; g < 16; ++g) {
      float4 s2 = rS[tid + 16*g], q2 = rQ[tid + 16*g], m2 = rM[tid + 16*g];
      S.x += s2.x; S.y += s2.y; S.z += s2.z; S.w += s2.w;
      Q.x += q2.x; Q.y += q2.y; Q.z += q2.z; Q.w += q2.w;
      M.x = fmaxf(M.x, m2.x); M.y = fmaxf(M.y, m2.y);
      M.z = fmaxf(M.z, m2.z); M.w = fmaxf(M.w, m2.w);
    }
    float* base = part + (size_t)(b * PB + pb) * 192;
    *(float4*)(base +       4*tid) = S;
    *(float4*)(base +  64 + 4*tid) = Q;
    *(float4*)(base + 128 + 4*tid) = M;
  }
}

// ---------------------------------------------------------------------------
// Kernel 2: degree scatter — integer atomics (order-independent => deterministic)
// ---------------------------------------------------------------------------
__global__ __launch_bounds__(256) void degree_kernel(const long long* __restrict__ dst,
                                                     unsigned* __restrict__ deg, int E) {
  int i = blockIdx.x * blockDim.x + threadIdx.x;
  int stride = gridDim.x * blockDim.x;
  for (; i < E; i += stride) {
    int idx = (int)dst[i];
    atomicAdd(&deg[idx], 1u);
  }
}

// ---------------------------------------------------------------------------
// Kernel 3: sum of squared degrees (exact integer accumulation, deterministic)
// ---------------------------------------------------------------------------
__global__ __launch_bounds__(256) void degsq_kernel(const unsigned* __restrict__ deg,
                                                    unsigned* __restrict__ sumsq, int N) {
  __shared__ unsigned red[256];
  const int tid = threadIdx.x;
  int i = blockIdx.x * 256 + tid;
  unsigned v = (i < N) ? deg[i] : 0u;
  red[tid] = v * v;
  __syncthreads();
  for (int s = 128; s > 0; s >>= 1) {
    if (tid < s) red[tid] += red[tid + s];
    __syncthreads();
  }
  if (tid == 0) atomicAdd(sumsq, red[0]);
}

// ---------------------------------------------------------------------------
// Kernel 4: final reduce + WMMA MLP. 1 block, 128 threads = 4 wave32s.
// Layer1: gf[16x196] @ W1[196x64]  -> 4 waves, one 16x16 N-tile each.
// Layer2: h1[16x64]  @ W2[64x32]   -> waves 0,1.
// Layer3 + sigmoid + means: scalar.
// ---------------------------------------------------------------------------
__global__ __launch_bounds__(128) void mlp_kernel(
    const float* __restrict__ part, const unsigned* __restrict__ degSS,
    const float* __restrict__ W1, const float* __restrict__ b1,
    const float* __restrict__ W2, const float* __restrict__ b2,
    const float* __restrict__ W3, const float* __restrict__ b3,
    float* __restrict__ out, int N, float avg_deg, double e2n,
    float density, float normsize) {
  __shared__ float gf[16][196];
  __shared__ float h1[16][64];
  __shared__ float h2[16][32];
  __shared__ float rawv[8];
#if !HAS_F32_WMMA
  __shared__ _Float16 gfH[16][224];   // K padded to 224 for K-step 32
  __shared__ _Float16 h1H[16][64];
#endif
  const int tid = threadIdx.x;

  // zero padding rows 8..15 of gf (rows 0..7 fully written below)
  for (int idx = tid; idx < 8 * 196; idx += 128)
    gf[8 + idx / 196][idx % 196] = 0.f;

  // reduce x-stats partials: 512 (b,d) pairs over PB partials each
  for (int p = tid; p < 512; p += 128) {
    int b = p >> 6, d = p & 63;
    float s = 0.f, q = 0.f, m = -INFINITY;
    const float* base = part + (size_t)b * PB * 192;
    for (int pb = 0; pb < PB; ++pb) {
      const float* pp = base + (size_t)pb * 192;
      s += pp[d];
      q += pp[64 + d];
      m = fmaxf(m, pp[128 + d]);
    }
    float mean = s / (float)N;
    float var  = fmaxf(q - s * s / (float)N, 0.f) / (float)(N - 1); // ddof=1
    gf[b][d]        = mean;
    gf[b][64 + d]   = m;
    gf[b][128 + d]  = sqrtf(var);
  }
  // structural features (same 4 values broadcast to all 8 rows)
  if (tid < 32) {
    int b = tid >> 2, c = tid & 3;
    double ssq  = (double)(*degSS);
    double vard = (ssq - e2n) / (double)(N - 1);
    float stdd  = (float)sqrt(vard > 0.0 ? vard : 0.0);
    float v = (c == 0) ? avg_deg : (c == 1) ? stdd : (c == 2) ? density : normsize;
    gf[b][192 + c] = v;
  }
#if !HAS_F32_WMMA
  __syncthreads();
  for (int idx = tid; idx < 16 * 224; idx += 128) {
    int r = idx / 224, cc = idx % 224;
    gfH[r][cc] = (cc < 196) ? (_Float16)gf[r][cc] : (_Float16)0.f;
  }
#endif
  __syncthreads();

  const int wave = tid >> 5;
  const int lane = tid & 31;
  const int half = lane >> 4;   // lane group: 0 -> lanes 0-15, 1 -> lanes 16-31
  const int l    = lane & 15;

  // ---- Layer 1: gf @ W1 + b1, relu ----
  {
    const int n0 = wave * 16;
    v8f c;
#pragma unroll
    for (int r = 0; r < 8; ++r) c[r] = b1[n0 + l];   // bias per column
#if HAS_F32_WMMA
    // f32 A 16x4: VGPR0 = K(+2*half), VGPR1 = K+1(+2*half). B 4x16 mirrors it.
    for (int k0 = 0; k0 < 196; k0 += 4) {
      const int ka = k0 + 2 * half;
      v2f a, bb;
      a[0]  = gf[l][ka];
      a[1]  = gf[l][ka + 1];
      bb[0] = W1[ka * 64 + n0 + l];
      bb[1] = W1[(ka + 1) * 64 + n0 + l];
      c = __builtin_amdgcn_wmma_f32_16x16x4_f32(false, a, false, bb,
                                                (short)0, c, false, false);
    }
#else
    for (int k0 = 0; k0 < 224; k0 += 32) {
      v16h a, bb;
#pragma unroll
      for (int r = 0; r < 4; ++r)
#pragma unroll
        for (int j = 0; j < 2; ++j) {
          a[2*r + j]     = gfH[l][k0 + 2*r + j + 8*half];
          a[8 + 2*r + j] = gfH[l][k0 + 16 + 2*r + j + 8*half];
        }
#pragma unroll
      for (int r = 0; r < 8; ++r)
#pragma unroll
        for (int j = 0; j < 2; ++j) {
          int K = k0 + 8*half + 2*r + j + (r >= 4 ? 8 : 0);
          bb[2*r + j] = (K < 196) ? (_Float16)W1[K * 64 + n0 + l] : (_Float16)0.f;
        }
      c = __builtin_amdgcn_wmma_f32_16x16x32_f16(false, a, false, bb,
                                                 (short)0, c, false, false);
    }
#endif
#pragma unroll
    for (int r = 0; r < 8; ++r) {
      float hv = fmaxf(c[r], 0.f);            // relu
      h1[r + 8*half][n0 + l] = hv;            // C/D layout: row r+8*half, col l
#if !HAS_F32_WMMA
      h1H[r + 8*half][n0 + l] = (_Float16)hv;
#endif
    }
  }
  __syncthreads();

  // ---- Layer 2: h1 @ W2 + b2, relu (waves 0,1; uniform branch => EXEC all 1s) ----
  if (wave < 2) {
    const int n0 = wave * 16;
    v8f c;
#pragma unroll
    for (int r = 0; r < 8; ++r) c[r] = b2[n0 + l];
#if HAS_F32_WMMA
    for (int k0 = 0; k0 < 64; k0 += 4) {
      const int ka = k0 + 2 * half;
      v2f a, bb;
      a[0]  = h1[l][ka];
      a[1]  = h1[l][ka + 1];
      bb[0] = W2[ka * 32 + n0 + l];
      bb[1] = W2[(ka + 1) * 32 + n0 + l];
      c = __builtin_amdgcn_wmma_f32_16x16x4_f32(false, a, false, bb,
                                                (short)0, c, false, false);
    }
#else
    for (int k0 = 0; k0 < 64; k0 += 32) {
      v16h a, bb;
#pragma unroll
      for (int r = 0; r < 4; ++r)
#pragma unroll
        for (int j = 0; j < 2; ++j) {
          a[2*r + j]     = h1H[l][k0 + 2*r + j + 8*half];
          a[8 + 2*r + j] = h1H[l][k0 + 16 + 2*r + j + 8*half];
        }
#pragma unroll
      for (int r = 0; r < 8; ++r)
#pragma unroll
        for (int j = 0; j < 2; ++j) {
          int K = k0 + 8*half + 2*r + j + (r >= 4 ? 8 : 0);
          bb[2*r + j] = (_Float16)W2[K * 32 + n0 + l];
        }
      c = __builtin_amdgcn_wmma_f32_16x16x32_f16(false, a, false, bb,
                                                 (short)0, c, false, false);
    }
#endif
#pragma unroll
    for (int r = 0; r < 8; ++r)
      h2[r + 8*half][n0 + l] = fmaxf(c[r], 0.f);
  }
  __syncthreads();

  // ---- Layer 3 + sigmoid ----
  if (tid < 8) {
    float z = b3[0];
#pragma unroll
    for (int j = 0; j < 32; ++j) z += h2[tid][j] * W3[j];
    rawv[tid] = 1.f / (1.f + expf(-z));
  }
  __syncthreads();

  if (tid == 0) {
    float sr = 0.f, sc = 0.f;
    for (int i = 0; i < 8; ++i) {
      float r = rawv[i];
      sr += r;
      float cont = 3.0f + r * 47.0f;          // MIN + raw*(MAX-MIN)
      sc += rintf(cont);                      // RNE == jnp.round (half-to-even)
    }
    out[0] = sc * 0.125f;   // mean(rounded)
    out[1] = sr * 0.125f;   // mean(raw)
  }
}

// ---------------------------------------------------------------------------
extern "C" void kernel_launch(void* const* d_in, const int* in_sizes, int n_in,
                              void* d_out, int out_size, void* d_ws, size_t ws_size,
                              hipStream_t stream) {
  const float*     x  = (const float*)d_in[0];
  const long long* ei = (const long long*)d_in[1];   // int64 [2, E]
  const float* W1 = (const float*)d_in[2];
  const float* b1 = (const float*)d_in[3];
  const float* W2 = (const float*)d_in[4];
  const float* b2 = (const float*)d_in[5];
  const float* W3 = (const float*)d_in[6];
  const float* b3 = (const float*)d_in[7];
  float* out = (float*)d_out;

  const int Bb     = 8;
  const int H      = in_sizes[3];            // 64
  const int in_dim = in_sizes[2] / H;        // 196
  const int D      = (in_dim - 4) / 3;       // 64
  const int N      = in_sizes[0] / (Bb * D); // 100000
  const int E      = in_sizes[1] / 2;        // 3200000

  // workspace: [deg: N u32][pad to 16: degSumSq u32][partials: B*PB*192 f32]
  char* ws = (char*)d_ws;
  size_t degBytes = (size_t)N * 4;
  unsigned* deg   = (unsigned*)ws;
  unsigned* degSS = (unsigned*)(ws + degBytes);
  float*    part  = (float*)(ws + degBytes + 16);

  hipMemsetAsync(d_ws, 0, degBytes + 16, stream);  // zero counters (capture-legal)

  const int RPB = (N + PB - 1) / PB;               // rows per block (400)
  dim3 gridX(PB, Bb);
  xstats_kernel<<<gridX, 256, 0, stream>>>(x, part, N, RPB);

  int degBlocks = (E + 1023) / 1024;
  degree_kernel<<<degBlocks, 256, 0, stream>>>(ei + E, deg, E);  // edge_index[1]

  int sqBlocks = (N + 255) / 256;
  degsq_kernel<<<sqBlocks, 256, 0, stream>>>(deg, degSS, N);

  double maxEdges = (double)N * (double)(N - 1) * 0.5;
  float density  = (float)((double)(E / 2) / maxEdges);
  float normsize = (float)(log((double)N + 1.0) / 10.0);
  float avg_deg  = (float)((double)E / (double)N);
  double e2n     = (double)E * (double)E / (double)N;

  mlp_kernel<<<1, 128, 0, stream>>>(part, degSS, W1, b1, W2, b2, W3, b3, out,
                                    N, avg_deg, e2n, density, normsize);
}
